// GraphConvolution_31550829756529
// MI455X (gfx1250) — matrix operand order repaired and checked by reading, not compiled, not run on hardware
//
#include <hip/hip_runtime.h>

#define NROWS 16384
#define DIN   512
#define DOUT  128
#define KEEP_SCALE (1.0f / 0.9f)

typedef __attribute__((ext_vector_type(16))) __bf16        v16bf;
typedef __attribute__((ext_vector_type(8)))  float         v8f;
typedef __attribute__((ext_vector_type(4)))  float         v4f;
typedef __attribute__((ext_vector_type(4)))  unsigned int  v4u;
typedef __attribute__((ext_vector_type(4)))  int           v4i;

#if __has_builtin(__builtin_amdgcn_global_load_async_to_lds_b128)
#define GC_ASYNC 1
#endif

#if __has_builtin(__builtin_amdgcn_sched_barrier)
#define GC_SCHED_FENCE() __builtin_amdgcn_sched_barrier(0)
#else
#define GC_SCHED_FENCE()
#endif

// pointer-to-v4i in global (AS1) / LDS (AS3) address spaces
typedef __attribute__((address_space(1))) v4i* gc_g128_t;
typedef __attribute__((address_space(3))) v4i* gc_l128_t;

// 16-byte global -> LDS copy; async on gfx1250, sync fallback otherwise.
__device__ __forceinline__ void gc_copy16(void* l, const void* g) {
#ifdef GC_ASYNC
  __builtin_amdgcn_global_load_async_to_lds_b128(
      (gc_g128_t)(unsigned long long)g,
      (gc_l128_t)(unsigned)(unsigned long long)l, 0, 0);
#else
  *(v4u*)l = *(const v4u*)g;
#endif
}

__device__ __forceinline__ void gc_async_wait() {
#ifdef GC_ASYNC
#if __has_builtin(__builtin_amdgcn_s_wait_asynccnt)
  __builtin_amdgcn_s_wait_asynccnt(0);
#else
  asm volatile("s_wait_asynccnt 0x0" ::: "memory");
#endif
#endif
}

// round-to-nearest-even f32 -> bf16
__device__ __forceinline__ unsigned short f2bf(float f) {
  union { float f; unsigned u; } x; x.f = f;
  unsigned u = x.u;
  u += 0x7FFFu + ((u >> 16) & 1u);
  return (unsigned short)(u >> 16);
}
__device__ __forceinline__ unsigned pack2(float lo, float hi) {
  return (unsigned)f2bf(lo) | ((unsigned)f2bf(hi) << 16);
}

// xd = bf16(x * mask * 1/(1-p)), 4 elements per thread
__global__ void gc_dropout_cast_kernel(const float* __restrict__ x,
                                       const float* __restrict__ mask,
                                       unsigned short* __restrict__ xd, int n4) {
  int i = blockIdx.x * blockDim.x + threadIdx.x;
  if (i >= n4) return;
  v4f xv = ((const v4f*)x)[i];
  v4f mv = ((const v4f*)mask)[i];
  unsigned lo = pack2(xv.x * mv.x * KEEP_SCALE, xv.y * mv.y * KEEP_SCALE);
  unsigned hi = pack2(xv.z * mv.z * KEEP_SCALE, xv.w * mv.w * KEEP_SCALE);
  ((unsigned long long*)xd)[i] =
      (unsigned long long)lo | ((unsigned long long)hi << 32);
}

// Wt[d][k] = bf16(W[k][d])
__global__ void gc_wt_cast_kernel(const float* __restrict__ w,
                                  unsigned short* __restrict__ wt) {
  int i = blockIdx.x * blockDim.x + threadIdx.x;
  if (i >= DIN * DOUT) return;
  int k = i / DOUT, d = i % DOUT;
  wt[d * DIN + k] = f2bf(w[i]);
}

// C[M,128] = A[M,K] @ B[K,128];  Bt = B^T row-major bf16 [128][K].
// A f32 (on-the-fly bf16 cast) or bf16. Double-buffered async LDS staging,
// KCHUNK=64 (16 wmma per barrier). 256 thr = 8 waves, 128-row M tile.
// sched_barriers pin the pipeline: preload all fragments, then interleave
// wmma + refill so no wmma waits on a dscnt==0.
template <bool A_F32, bool FINAL>
__global__ __launch_bounds__(256) void
gc_gemm_wmma_kernel(const void* __restrict__ Aptr, int lda,
                    const unsigned short* __restrict__ Bt, int K,
                    unsigned short* __restrict__ Ct,
                    float* __restrict__ out) {
  constexpr int KCHUNK   = 64;
  constexpr int A_STRIDE = A_F32 ? 68 : 72;              // elements, padded
  constexpr int A_ELT    = A_F32 ? 4 : 2;                // bytes
  constexpr int A_BUF    = 128 * A_STRIDE * A_ELT;       // bytes per buffer
  constexpr int B_STRIDE = 72;                           // bf16 elements
  constexpr int B_BUF    = 128 * B_STRIDE * 2;           // bytes per buffer

  __shared__ __align__(16) unsigned char smem[2 * A_BUF + 2 * B_BUF];
  unsigned char* Asm = smem;
  unsigned char* Bsm = smem + 2 * A_BUF;

  const int t    = threadIdx.x;
  const int lane = t & 31;
  const int wave = t >> 5;
  const int half = lane >> 4;
  const int nidx = lane & 15;
  const int m0   = blockIdx.x * 128;
  const int arow = wave * 16 + nidx;

  const float*          Af = (const float*)Aptr;
  const unsigned short* Ab = (const unsigned short*)Aptr;

  // cooperative staging: thread t covers (row = t>>1, part = t&1)
  const int srow  = t >> 1;
  const int spart = t & 1;

  auto stage = [&](int buf, int k0) {
    {
      const unsigned short* g = Bt + (size_t)srow * K + k0 + spart * 32;
      unsigned char* l = Bsm + buf * B_BUF + (srow * B_STRIDE + spart * 32) * 2;
#pragma unroll
      for (int j = 0; j < 4; ++j) gc_copy16(l + j * 16, g + j * 8);
    }
    if (A_F32) {
      const float* g = Af + (size_t)(m0 + srow) * lda + k0 + spart * 32;
      unsigned char* l = Asm + buf * A_BUF + (srow * A_STRIDE + spart * 32) * 4;
#pragma unroll
      for (int j = 0; j < 8; ++j) gc_copy16(l + j * 16, g + j * 4);
    } else {
      const unsigned short* g = Ab + (size_t)(m0 + srow) * lda + k0 + spart * 32;
      unsigned char* l = Asm + buf * A_BUF + (srow * A_STRIDE + spart * 32) * 2;
#pragma unroll
      for (int j = 0; j < 4; ++j) gc_copy16(l + j * 16, g + j * 8);
    }
  };

  v8f acc[8];
#pragma unroll
  for (int nt = 0; nt < 8; ++nt)
#pragma unroll
    for (int r = 0; r < 8; ++r) acc[nt][r] = 0.0f;

  const int nchunks = K / KCHUNK;
  stage(0, 0);
  gc_async_wait();
  __syncthreads();

  for (int c = 0; c < nchunks; ++c) {
    const int buf = c & 1;
    if (c + 1 < nchunks) {
      stage(buf ^ 1, (c + 1) * KCHUNK);  // long-latency HBM async copies first
      GC_SCHED_FENCE();
    }

    const unsigned char* Ab_lds = Asm + buf * A_BUF;
    const unsigned char* Bb_lds = Bsm + buf * B_BUF;

    union AF { v16bf v; unsigned u[8]; v4u q[2]; };
    union BF { v16bf v; v4u q[2]; };
    AF  af[2];
    BF  bf[8];           // all 8 N-tiles of one k-step live at once
    v4f araw[2][4];

    auto issueA = [&](int s) {
      if (A_F32) {
        const float* ap =
            (const float*)Ab_lds + arow * A_STRIDE + s * 32 + half * 8;
        araw[s][0] = *(const v4f*)(ap);
        araw[s][1] = *(const v4f*)(ap + 4);
        araw[s][2] = *(const v4f*)(ap + 16);
        araw[s][3] = *(const v4f*)(ap + 20);
      } else {
        const unsigned short* ap =
            (const unsigned short*)Ab_lds + arow * A_STRIDE + s * 32 + half * 8;
        af[s].q[0] = *(const v4u*)(ap);
        af[s].q[1] = *(const v4u*)(ap + 16);
      }
    };
    auto convA = [&](int s) {  // f32 -> bf16 fragment pack (no-op for bf16 A)
      if (A_F32) {
        af[s].u[0] = pack2(araw[s][0].x, araw[s][0].y);
        af[s].u[1] = pack2(araw[s][0].z, araw[s][0].w);
        af[s].u[2] = pack2(araw[s][1].x, araw[s][1].y);
        af[s].u[3] = pack2(araw[s][1].z, araw[s][1].w);
        af[s].u[4] = pack2(araw[s][2].x, araw[s][2].y);
        af[s].u[5] = pack2(araw[s][2].z, araw[s][2].w);
        af[s].u[6] = pack2(araw[s][3].x, araw[s][3].y);
        af[s].u[7] = pack2(araw[s][3].z, araw[s][3].w);
      }
    };
    auto loadB = [&](int nt, int s) {
      const unsigned short* bp = (const unsigned short*)Bb_lds +
                                 (nt * 16 + nidx) * B_STRIDE + s * 32 +
                                 half * 8;
      bf[nt].q[0] = *(const v4u*)(bp);
      bf[nt].q[1] = *(const v4u*)(bp + 16);
    };

    // Region 1: issue ALL fragment loads for k-step 0 (+ A for k-step 1),
    // then pack A. Nothing may sink past the fence, so all 8 B fragments
    // stay simultaneously live in distinct registers.
    issueA(0);
    issueA(1);
#pragma unroll
    for (int nt = 0; nt < 8; ++nt) loadB(nt, 0);
    convA(0);
    convA(1);
    GC_SCHED_FENCE();

    // k-step 0: wmma(nt) then refill bf[nt] for k-step 1; fences force the
    // interleave so each refill is issued 8 WMMAs before its consumer.
#pragma unroll
    for (int nt = 0; nt < 8; ++nt) {
      acc[nt] = __builtin_amdgcn_wmma_f32_16x16x32_bf16(
          false, af[0].v, false, bf[nt].v, (short)0, acc[nt], false, false);
      loadB(nt, 1);
      GC_SCHED_FENCE();
    }

    // k-step 1: pure WMMA drain
#pragma unroll
    for (int nt = 0; nt < 8; ++nt) {
      acc[nt] = __builtin_amdgcn_wmma_f32_16x16x32_bf16(
          false, af[1].v, false, bf[nt].v, (short)0, acc[nt], false, false);
    }

    gc_async_wait();
    __syncthreads();
  }

  const int mw = m0 + wave * 16;
  if (FINAL) {
#pragma unroll
    for (int nt = 0; nt < 8; ++nt) {
      int col = nt * 16 + nidx;
#pragma unroll
      for (int r = 0; r < 8; ++r) {
        float v = acc[nt][r];
        out[(size_t)(mw + half * 8 + r) * DOUT + col] = v > 0.0f ? v : 0.0f;
      }
    }
  } else {
#pragma unroll
    for (int nt = 0; nt < 8; ++nt) {
      int col = nt * 16 + nidx;
      union { v4u q; unsigned short s[8]; } pk;
#pragma unroll
      for (int r = 0; r < 8; ++r) pk.s[r] = f2bf(acc[nt][r]);
      *(v4u*)(Ct + (size_t)col * NROWS + mw + half * 8) = pk.q;
    }
  }
}

extern "C" void kernel_launch(void* const* d_in, const int* in_sizes, int n_in,
                              void* d_out, int out_size, void* d_ws, size_t ws_size,
                              hipStream_t stream) {
  const float* x       = (const float*)d_in[0];
  const float* support = (const float*)d_in[1];
  const float* weight  = (const float*)d_in[2];
  const float* mask    = (const float*)d_in[3];
  float*       out     = (float*)d_out;

  char* ws = (char*)d_ws;
  unsigned short* xd = (unsigned short*)ws;                               // 16 MB
  unsigned short* wt = (unsigned short*)(ws + (size_t)NROWS * DIN * 2);   // 128 KB
  unsigned short* fA = (unsigned short*)(ws + (size_t)NROWS * DIN * 2 +
                                         (size_t)DIN * DOUT * 2);         // 4 MB
  unsigned short* fB = fA + (size_t)DOUT * NROWS;                         // 4 MB

  int n4 = NROWS * DIN / 4;
  gc_dropout_cast_kernel<<<(n4 + 255) / 256, 256, 0, stream>>>(x, mask, xd, n4);
  gc_wt_cast_kernel<<<(DIN * DOUT + 255) / 256, 256, 0, stream>>>(weight, wt);

  // pre_sup^T = (xd @ W)^T   (K = 512)
  gc_gemm_wmma_kernel<false, false><<<NROWS / 128, 256, 0, stream>>>(
      xd, DIN, wt, DIN, fA, nullptr);

  // three diffusion steps feat <- support @ feat (K = 16384), last fused ReLU
  gc_gemm_wmma_kernel<true, false><<<NROWS / 128, 256, 0, stream>>>(
      support, NROWS, fA, NROWS, fB, nullptr);
  gc_gemm_wmma_kernel<true, false><<<NROWS / 128, 256, 0, stream>>>(
      support, NROWS, fB, NROWS, fA, nullptr);
  gc_gemm_wmma_kernel<true, true><<<NROWS / 128, 256, 0, stream>>>(
      support, NROWS, fA, NROWS, nullptr, out);
}
